// FusedAttention_48610439856667
// MI455X (gfx1250) — matrix operand order repaired and checked by reading, not compiled, not run on hardware
//
#include <hip/hip_runtime.h>
#include <hip/hip_bf16.h>

// ---------------- problem constants (from reference) ----------------
#define B_      32
#define HIDDEN  4096
#define NH      32
#define NKV     8
#define HD      128
#define PAST    2048
#define KEYS    2049          // PAST + 1 new token
#define NSPLIT  8             // key-dimension splits for attention
#define SLK     288           // keys per slice: 8*288 = 2304 >= 2049; 288 = 9*32
#define QKVCOLS 6144          // 4096 q + 1024 k + 1024 v

typedef __attribute__((ext_vector_type(16))) _Float16 v16h;
typedef __attribute__((ext_vector_type(8)))  _Float16 v8h;
typedef __attribute__((ext_vector_type(8)))  float    v8f;

// CDNA5 WMMA f32 = f16(16x32) x f16(32x16) + f32
__device__ __forceinline__ v8f wmma_f16(v16h a, v16h b, v8f c) {
    return __builtin_amdgcn_wmma_f32_16x16x32_f16(
        false, a, false, b, (short)0, c, false, false);
}

// A-operand (16x32 f16): lane%16 = row, half h holds K = (lane>>4)*16 + h
// -> two contiguous 16-byte chunks at K-offsets kb and kb+8.
__device__ __forceinline__ v16h load_a16(const _Float16* p) {
    v16h a;
    *(v8h*)&a       = *(const v8h*)(p);
    *((v8h*)&a + 1) = *(const v8h*)(p + 8);
    return a;
}

__device__ __forceinline__ v16h zero16h() {
    v16h z;
#pragma unroll
    for (int i = 0; i < 16; ++i) z[i] = (_Float16)0.0f;
    return z;
}

// Non-temporal 32-byte load (KV cache is a read-once 268MB stream > 192MB L2:
// keep it out of the caches so weight reuse survives).
__device__ __forceinline__ v16h load_b16_nt(const _Float16* p) {
    v16h b;
    *(v8h*)&b       = __builtin_nontemporal_load((const v8h*)(p));
    *((v8h*)&b + 1) = __builtin_nontemporal_load((const v8h*)(p + 8));
    return b;
}

// =====================================================================
// Kernel 1: fused QKV projection, split-K by 2, f32 partials.
// grid = 2 splits * (2 row tiles * 384 col tiles); one wave per 16x16 tile.
// partial layout: part[s][row(32)][QKVCOLS]  (cols: 0..4095 q, +1024 k, +1024 v)
// =====================================================================
__global__ __launch_bounds__(32) void qkv_gemm(
    const _Float16* __restrict__ x,
    const _Float16* __restrict__ wq,
    const _Float16* __restrict__ wk,
    const _Float16* __restrict__ wv,
    float* __restrict__ part)
{
    const int s    = blockIdx.x / 768;
    const int tile = blockIdx.x - s * 768;
    const int rt   = tile / 384;
    const int ct   = tile - rt * 384;

    const _Float16* W; int wcb, gcb;
    if (ct < 256)      { W = wq; wcb = ct * 16;        gcb = wcb;        }
    else if (ct < 320) { W = wk; wcb = (ct - 256) * 16; gcb = 4096 + wcb; }
    else               { W = wv; wcb = (ct - 320) * 16; gcb = 5120 + wcb; }

    const int lane = threadIdx.x & 31;
    const int l15  = lane & 15;
    const int hi   = lane >> 4;
    const int kb   = s * 2048;

    const _Float16* ap = x + (size_t)(rt * 16 + l15) * HIDDEN + kb + hi * 16;
    const _Float16* bp = W + (size_t)(wcb + l15) * HIDDEN + kb + hi * 16;

    v8f acc = {};
    for (int kk = 0; kk < 2048; kk += 32) {
        __builtin_prefetch(bp + kk + 1024, 0, 0);   // global_prefetch_b8
        v16h a = load_a16(ap + kk);
        v16h b = *(const v16h*)(bp + kk);
        acc = wmma_f16(a, b, acc);
    }
    const int mrow = rt * 16 + hi * 8;
#pragma unroll
    for (int v = 0; v < 8; ++v)
        part[(size_t)(s * 32 + mrow + v) * QKVCOLS + gcb + l15] = acc[v];
}

// =====================================================================
// Kernel 2: RoPE fused with the split-K reduction of the q/k partials.
// =====================================================================
__global__ __launch_bounds__(256) void rope_kernel(
    const float* __restrict__ part,      // [2][32][QKVCOLS]
    const _Float16* __restrict__ cosp, const _Float16* __restrict__ sinp,
    _Float16* __restrict__ qo, _Float16* __restrict__ ko)
{
    int idx = blockIdx.x * blockDim.x + threadIdx.x;
    if (idx >= B_ * (NH + NKV) * HD) return;
    const int d = idx & 127;
    const int h = (idx >> 7) % (NH + NKV);
    const int b = idx / ((NH + NKV) * HD);
    const int dp = (d < 64) ? d + 64 : d - 64;

    const int colA = (h < NH) ? h * HD + d  : 4096 + (h - NH) * HD + d;
    const int colR = (h < NH) ? h * HD + dp : 4096 + (h - NH) * HD + dp;

    const float t  = part[(size_t)b * QKVCOLS + colA] +
                     part[(size_t)(32 + b) * QKVCOLS + colA];
    float rot      = part[(size_t)b * QKVCOLS + colR] +
                     part[(size_t)(32 + b) * QKVCOLS + colR];
    if (d < 64) rot = -rot;

    const float c = (float)cosp[b * HD + d];
    const float s = (float)sinp[b * HD + d];
    const float val = t * c + rot * s;
    if (h < NH) qo[(size_t)b * 4096 + h * HD + d]        = (_Float16)val;
    else        ko[(size_t)b * 1024 + (h - NH) * HD + d] = (_Float16)val;
}

// Kernel 2b: reduce V partials to f16.
__global__ __launch_bounds__(256) void reduce_v(
    const float* __restrict__ part, _Float16* __restrict__ vws)
{
    int i = blockIdx.x * blockDim.x + threadIdx.x;
    if (i >= B_ * 1024) return;
    const int b = i >> 10, c = i & 1023;
    const int col = 5120 + c;
    vws[i] = (_Float16)(part[(size_t)b * QKVCOLS + col] +
                        part[(size_t)(32 + b) * QKVCOLS + col]);
}

// =====================================================================
// Kernel 3: attention partials. One wave per (b, g, key-slice s).
// grid = 32*8*8 = 2048 blocks -> 8x the KV-stream memory parallelism.
// Produces unnormalized o_s = sum e^{x-m_s} v, plus (m_s, s_s) per head.
// =====================================================================
__global__ __launch_bounds__(32) void attn_part(
    const _Float16* __restrict__ qrot,   // [32][4096]
    const _Float16* __restrict__ krot,   // [32][1024]
    const _Float16* __restrict__ vnew,   // [32][1024]
    const _Float16* __restrict__ kvbuf,  // [SLOTS][16][128]
    const int*      __restrict__ sel,    // [32*2048]
    float* __restrict__ pout,            // [2048][4][128]
    float* __restrict__ pm,              // [2048][4]
    float* __restrict__ ps)              // [2048][4]
{
    __shared__ float    sc[4 * SLK];
    __shared__ _Float16 vstage[32 * HD];
    __shared__ float    rowm[4], rowsum[4];

    const int idx = blockIdx.x;          // ((b*8+g)*8+s)
    const int s   = idx & 7;
    const int g   = (idx >> 3) & 7;
    const int b   = idx >> 6;
    const int k0  = s * SLK;
    const int lane = threadIdx.x & 31;
    const int l15  = lane & 15;
    const int hi   = lane >> 4;
    const float scale = 0.08838834764831845f;   // 1/sqrt(128)

    // rows 4..15 of A are clamped duplicates of head 3; D rows >= 4 are
    // never stored, and each D row depends only on its own A row.
    const int rowc = (l15 < 4) ? l15 : 3;
    const _Float16* qrow = qrot + (size_t)b * 4096 + (g * 4 + rowc) * HD + hi * 16;
    v16h aq[4];
#pragma unroll
    for (int kd = 0; kd < 4; ++kd) aq[kd] = load_a16(qrow + kd * 32);

    // ---------------- pass 1: slice scores -> LDS ----------------
    for (int c = 0; c < SLK / 16; ++c) {
        const int key = k0 + c * 16 + l15;          // this lane's key column
        const _Float16* kvec = nullptr;
        bool nt = false;                             // kv_buffer => non-temporal
        if (key < PAST) {
            kvec = kvbuf + (size_t)sel[b * PAST + key] * (2 * NKV * HD) + g * HD + hi * 16;
            nt = true;
        } else if (key == PAST) {
            kvec = krot + (size_t)b * 1024 + g * HD + hi * 16;
        }

        v8f acc = {};
#pragma unroll
        for (int kd = 0; kd < 4; ++kd) {
            v16h bm = zero16h();
            if (kvec) bm = nt ? load_b16_nt(kvec + kd * 32)       // lane-masked
                             : *(const v16h*)(kvec + kd * 32);
            acc = wmma_f16(aq[kd], bm, acc);                      // EXEC all-1s
        }
        if (hi == 0) {
#pragma unroll
            for (int v = 0; v < 4; ++v)
                sc[v * SLK + c * 16 + l15] = (key < KEYS) ? acc[v] * scale : -1.0e30f;
        }
    }
    __syncthreads();

    // ---------------- slice softmax stats ----------------
    for (int r = 0; r < 4; ++r) {
        float mx = -3.0e38f;
        for (int t = lane; t < SLK; t += 32) mx = fmaxf(mx, sc[r * SLK + t]);
#pragma unroll
        for (int sh = 16; sh > 0; sh >>= 1) mx = fmaxf(mx, __shfl_xor(mx, sh, 32));
        float sm = 0.0f;
        for (int t = lane; t < SLK; t += 32) sm += __expf(sc[r * SLK + t] - mx);
#pragma unroll
        for (int sh = 16; sh > 0; sh >>= 1) sm += __shfl_xor(sm, sh, 32);
        if (lane == 0) { rowm[r] = mx; rowsum[r] = sm; }
    }
    __syncthreads();
    const float mrow = rowm[rowc];

    // ---------------- pass 2: o_s = P~ * V  (unnormalized) ----------------
    v8f zz = {};
    v8f accO[8];
#pragma unroll
    for (int i = 0; i < 8; ++i) accO[i] = zz;

    for (int kc = 0; kc < SLK; kc += 32) {
        const int key = k0 + kc + lane;             // one V row per lane
        const _Float16* vptr = nullptr;
        bool nt = false;
        if (key < PAST) {
            vptr = kvbuf + (size_t)sel[b * PAST + key] * (2 * NKV * HD) + (NKV + g) * HD;
            nt = true;
        } else if (key == PAST) {
            vptr = vnew + (size_t)b * 1024 + g * HD;
        }

        __syncthreads();                            // prior chunk's reads done
        if (vptr) {
            if (nt) {
#pragma unroll
                for (int j = 0; j < HD; j += 8)
                    *(v8h*)&vstage[lane * HD + j] =
                        __builtin_nontemporal_load((const v8h*)(vptr + j));
            } else {
#pragma unroll
                for (int j = 0; j < HD; j += 8)
                    *(v8h*)&vstage[lane * HD + j] = *(const v8h*)(vptr + j);
            }
        } else {
            v8h zh = {};
#pragma unroll
            for (int j = 0; j < HD; j += 8)
                *(v8h*)&vstage[lane * HD + j] = zh;
        }
        __syncthreads();

        v16h a;
#pragma unroll
        for (int j = 0; j < 16; ++j) {
            const int lk = kc + hi * 16 + j;
            a[j] = (_Float16)__expf(sc[rowc * SLK + lk] - mrow);  // 0 if masked
        }
#pragma unroll
        for (int dt = 0; dt < 8; ++dt) {            // 8 x 16 dims = 128
            v16h bm;
#pragma unroll
            for (int j = 0; j < 16; ++j)
                bm[j] = vstage[(hi * 16 + j) * HD + dt * 16 + l15];
            accO[dt] = wmma_f16(a, bm, accO[dt]);
        }
    }

    if (hi == 0) {
#pragma unroll
        for (int dt = 0; dt < 8; ++dt)
#pragma unroll
            for (int v = 0; v < 4; ++v)
                pout[((size_t)idx * 4 + v) * HD + dt * 16 + l15] = accO[dt][v];
    }
    if (lane < 4) { pm[idx * 4 + lane] = rowm[lane]; ps[idx * 4 + lane] = rowsum[lane]; }
}

// =====================================================================
// Kernel 3b: merge slice partials.  out = sum_s o_s e^{m_s-M} / sum_s s_s e^{m_s-M}
// =====================================================================
__global__ __launch_bounds__(256) void attn_merge(
    const float* __restrict__ pout, const float* __restrict__ pm,
    const float* __restrict__ ps, _Float16* __restrict__ attn)
{
    int i = blockIdx.x * blockDim.x + threadIdx.x;
    if (i >= B_ * NH * HD) return;
    const int d = i & 127;
    const int h = (i >> 7) & 31;
    const int b = i >> 12;
    const int g = h >> 2, r = h & 3;
    const int base = (b * 8 + g) * 8;

    float m_[NSPLIT];
    float M = -3.0e38f;
#pragma unroll
    for (int s = 0; s < NSPLIT; ++s) {
        m_[s] = pm[(base + s) * 4 + r];
        M = fmaxf(M, m_[s]);
    }
    float den = 0.0f, num = 0.0f;
#pragma unroll
    for (int s = 0; s < NSPLIT; ++s) {
        const float w = __expf(m_[s] - M);
        den += ps[(base + s) * 4 + r] * w;
        num += pout[((size_t)(base + s) * 4 + r) * HD + d] * w;
    }
    attn[(size_t)b * 4096 + h * HD + d] = (_Float16)(num / den);
}

// =====================================================================
// Kernel 4: output projection, split-K by 2, f32 partials.
// wo is [o][h] row-major (N-contiguous) -> stage 32x16 chips via LDS.
// =====================================================================
__global__ __launch_bounds__(32) void oproj_gemm(
    const _Float16* __restrict__ attn,
    const _Float16* __restrict__ wo,
    float* __restrict__ part)            // [2][32][4096]
{
    __shared__ _Float16 wt[32 * 16];     // [k within chunk][n]

    const int s    = blockIdx.x / 512;
    const int rest = blockIdx.x - s * 512;
    const int rt   = rest >> 8;
    const int ct   = rest & 255;
    const int lane = threadIdx.x & 31;
    const int l15  = lane & 15, hi = lane >> 4;
    const int cb   = ct * 16;
    const int kb   = s * 2048;

    const _Float16* ap = attn + (size_t)(rt * 16 + l15) * 4096 + kb + hi * 16;

    v8f acc = {};
    for (int kk = 0; kk < 2048; kk += 32) {
        __syncthreads();
        const _Float16* wrow = wo + (size_t)(kb + kk + lane) * 4096 + cb;
        *(v8h*)&wt[lane * 16]     = *(const v8h*)(wrow);
        *(v8h*)&wt[lane * 16 + 8] = *(const v8h*)(wrow + 8);
        __syncthreads();

        v16h a = load_a16(ap + kk);
        v16h bm;
#pragma unroll
        for (int j = 0; j < 16; ++j)
            bm[j] = wt[(hi * 16 + j) * 16 + l15];
        acc = wmma_f16(a, bm, acc);
    }
#pragma unroll
    for (int v = 0; v < 8; ++v)
        part[(size_t)(s * 32 + rt * 16 + hi * 8 + v) * 4096 + cb + l15] = acc[v];
}

// Kernel 4b: reduce output-projection partials to the f16 result.
__global__ __launch_bounds__(256) void oreduce(
    const float* __restrict__ part, _Float16* __restrict__ out)
{
    int i = blockIdx.x * blockDim.x + threadIdx.x;
    if (i >= B_ * 4096) return;
    out[i] = (_Float16)(part[i] + part[(size_t)32 * 4096 + i]);
}

// =====================================================================
extern "C" void kernel_launch(void* const* d_in, const int* in_sizes, int n_in,
                              void* d_out, int out_size, void* d_ws, size_t ws_size,
                              hipStream_t stream) {
    (void)in_sizes; (void)n_in; (void)out_size; (void)ws_size;
    const _Float16* x     = (const _Float16*)d_in[0];
    const _Float16* wq    = (const _Float16*)d_in[1];
    const _Float16* wk    = (const _Float16*)d_in[2];
    const _Float16* wv    = (const _Float16*)d_in[3];
    const _Float16* wo    = (const _Float16*)d_in[4];
    const _Float16* cosp  = (const _Float16*)d_in[5];
    const _Float16* sinp  = (const _Float16*)d_in[6];
    const _Float16* kvbuf = (const _Float16*)d_in[7];
    const int*      sel   = (const int*)d_in[8];
    // d_in[9] = decode_index: unused by the reference computation

    _Float16* out = (_Float16*)d_out;

    // ---- carve workspace: f32 partials first, then f16 buffers ----
    float* fws     = (float*)d_ws;
    float* qkvpart = fws;                                  // 2*32*6144
    float* opart   = qkvpart + (size_t)2 * 32 * QKVCOLS;   // 2*32*4096
    float* pout    = opart   + (size_t)2 * 32 * 4096;      // 2048*4*128
    float* pm      = pout    + (size_t)2048 * 4 * HD;      // 2048*4
    float* ps      = pm      + (size_t)2048 * 4;           // 2048*4
    _Float16* hws  = (_Float16*)(ps + (size_t)2048 * 4);
    _Float16* qrot = hws;                                  // 32*4096
    _Float16* krot = qrot + (size_t)B_ * 4096;             // 32*1024
    _Float16* vws  = krot + (size_t)B_ * 1024;             // 32*1024
    _Float16* attn = vws  + (size_t)B_ * 1024;             // 32*4096

    qkv_gemm<<<2 * 768, 32, 0, stream>>>(x, wq, wk, wv, qkvpart);
    rope_kernel<<<(B_ * (NH + NKV) * HD + 255) / 256, 256, 0, stream>>>(
        qkvpart, cosp, sinp, qrot, krot);
    reduce_v<<<(B_ * 1024 + 255) / 256, 256, 0, stream>>>(qkvpart, vws);
    attn_part<<<B_ * NKV * NSPLIT, 32, 0, stream>>>(
        qrot, krot, vws, kvbuf, sel, pout, pm, ps);
    attn_merge<<<(B_ * NH * HD + 255) / 256, 256, 0, stream>>>(pout, pm, ps, attn);
    oproj_gemm<<<2 * 512, 32, 0, stream>>>(attn, wo, opart);
    oreduce<<<(B_ * 4096 + 255) / 256, 256, 0, stream>>>(opart, out);
}
